// VelocityHead_16114717294957
// MI455X (gfx1250) — matrix operand order repaired and checked by reading, not compiled, not run on hardware
//
#include <hip/hip_runtime.h>
#include <hip/hip_bf16.h>
#include <math.h>

typedef __attribute__((ext_vector_type(16))) _Float16 v16h;
typedef __attribute__((ext_vector_type(4)))  _Float16 v4h;
typedef __attribute__((ext_vector_type(8)))  float    v8f;
typedef __attribute__((ext_vector_type(4)))  float    v4f;

#define CCH 128
#define TDIM 64
#define HID 128

// ---------------------------------------------------------------------------
// Kernel 1: time-MLP  ->  per-batch bias table tb[B,128] and gate table tg[B,128]
// t_emb = [sin(t*f), cos(t*f)], hmid = silu(t_emb @ w1^T + b1),
// p = hmid @ w2^T + b2 ; tb = p[:128], tg = 1 + p[128:]
// ---------------------------------------------------------------------------
__global__ void tmlp_kernel(const float* __restrict__ t,
                            const float* __restrict__ w1, const float* __restrict__ b1,
                            const float* __restrict__ w2, const float* __restrict__ b2,
                            float* __restrict__ tb, float* __restrict__ tg, int B)
{
    __shared__ float temb[TDIM];
    __shared__ float hmid[HID];
    int b = blockIdx.x;
    if (b >= B) return;
    int i = threadIdx.x;            // 0..127
    float tv = t[b];
    if (i < TDIM) {
        int half = TDIM / 2;
        int f = i & (half - 1);
        float freq = expf(-logf(10000.f) * (float)f / (float)half);
        float arg = tv * freq;
        temb[i] = (i < half) ? sinf(arg) : cosf(arg);
    }
    __syncthreads();
    {
        float acc = b1[i];
        const float* wr = w1 + i * TDIM;
        for (int d = 0; d < TDIM; ++d) acc += temb[d] * wr[d];
        hmid[i] = acc / (1.f + expf(-acc));   // silu
    }
    __syncthreads();
    for (int o = 0; o < 2; ++o) {
        int k = i + o * HID;
        float acc = b2[k];
        const float* wr = w2 + k * HID;
        for (int d = 0; d < HID; ++d) acc += hmid[d] * wr[d];
        if (o == 0) tb[b * CCH + k] = acc;
        else        tg[b * CCH + (k - HID)] = 1.f + acc;
    }
}

// ---------------------------------------------------------------------------
// Kernel 2: pack lw0, lw1, gw (f32, 128x128, o-major) into f16 WMMA B-fragment
// layout.  Flat index = mat*16384 + kb*4096 + jb*512 + lane*16 + h.
// B[k][n] = W[n][k] with n = jb*16 + lane%16,
// k = kb*32 + (lane<16 ? 0 : 8) + (h<8 ? h : h+8)   (table-order K B layout, matches A)
// ---------------------------------------------------------------------------
__global__ void pack_kernel(const float* __restrict__ lw, const float* __restrict__ gw,
                            _Float16* __restrict__ wpack)
{
    int tid = blockIdx.x * blockDim.x + threadIdx.x;
    if (tid >= 3 * CCH * CCH) return;
    int h    =  tid        & 15;
    int lane = (tid >> 4)  & 31;
    int jb   = (tid >> 9)  & 7;
    int kb   = (tid >> 12) & 3;
    int mat  =  tid >> 14;
    int n = jb * 16 + (lane & 15);
    int k = kb * 32 + ((lane < 16) ? 0 : 8) + ((h < 8) ? h : h + 8);
    float v = (mat == 2) ? gw[n * CCH + k] : lw[mat * CCH * CCH + n * CCH + k];
    wpack[tid] = (_Float16)v;
}

// ---------------------------------------------------------------------------
// Main kernel helpers
// ---------------------------------------------------------------------------
// A fragment (16x32 f16) from a row-major 16x128 f16 LDS region.
// ISA A layout: lane<16 -> row=lane, K = kb*32 + {0..7, 16..23}
//               lane>=16 -> row=lane-16, K = kb*32 + {8..15, 24..31}
__device__ __forceinline__ v16h load_a_frag(const _Float16* base, int lane, int kb)
{
    int m  = lane & 15;
    int k0 = kb * 32 + ((lane < 16) ? 0 : 8);
    const _Float16* p = base + m * CCH + k0;
    v16h a;
#pragma unroll
    for (int i = 0; i < 8; ++i) { a[i] = p[i]; a[i + 8] = p[i + 16]; }
    return a;
}

__device__ __forceinline__ v16h load_b_frag(const _Float16* __restrict__ wpack,
                                            int mat, int kb, int jb, int lane)
{
    return *(const v16h*)(wpack + ((((mat * 4 + kb) * 8 + jb) * 32 + lane) << 4));
}

__device__ __forceinline__ v8f wmma_f16(v16h a, v16h b, v8f c)
{
    return __builtin_amdgcn_wmma_f32_16x16x32_f16(false, a, false, b, (short)0, c,
                                                  false, false);
}

// ---------------------------------------------------------------------------
// Main kernel: 128 threads = 4 waves, one 16-point tile per wave.
// ---------------------------------------------------------------------------
__global__ void __launch_bounds__(128)
velocity_head_kernel(const float* __restrict__ x, const int* __restrict__ batch_idx,
                     const float* __restrict__ lb, const float* __restrict__ gb,
                     const float* __restrict__ fw,
                     const _Float16* __restrict__ wpack,
                     const float* __restrict__ tb, const float* __restrict__ tg,
                     float* __restrict__ out, int N)
{
    extern __shared__ char smem[];
    const int wid  = threadIdx.x >> 5;
    const int lane = threadIdx.x & 31;
    _Float16* xg  = (_Float16*)smem + wid * (4 * 16 * CCH);         // 16KB/wave
    float*    w1g = (float*)(smem + 65536) + wid * (8 * 8 * 32);     // 8KB/wave

    const int ntiles = (N + 15) >> 4;
    const int tile = blockIdx.x * 4 + wid;
    if (tile >= ntiles) return;
    const int n0 = tile * 16;

    // ---- stage 1: stage adjusted inputs into LDS (f16, row-major per l) ----
    const int c4 = lane * 4;                    // 32 lanes x 4 floats = 128 c
    for (int m = 0; m < 16; ++m) {
        const int n = n0 + m;
        const bool valid = (n < N);             // uniform across the wave
        const int bidx = valid ? batch_idx[n] : 0;
        const float* xr = x + (size_t)n * (9 * CCH);
        v4f bv = *(const v4f*)(tb + bidx * CCH + c4);
        v4f gv = *(const v4f*)(tg + bidx * CCH + c4);
#pragma unroll
        for (int l = 0; l < 4; ++l) {
            v4f xv;
            if (valid) xv = *(const v4f*)(xr + l * CCH + c4);
            else { xv[0] = xv[1] = xv[2] = xv[3] = 0.f; }
            v4f y = (l == 0) ? (xv + bv) : (xv * gv);
            v4h hy;
            hy[0] = (_Float16)y[0]; hy[1] = (_Float16)y[1];
            hy[2] = (_Float16)y[2]; hy[3] = (_Float16)y[3];
            *(v4h*)(xg + l * (16 * CCH) + m * CCH + c4) = hy;
        }
    }
    // Intra-wave LDS ops are in-order; no barrier needed (regions are per-wave).

    // ---- stage 2: o0 = (x0+bias) @ lw0^T + lb ; write o0 (f16) over xg[l=0] ----
    v16h a0[4];
#pragma unroll
    for (int kb = 0; kb < 4; ++kb) a0[kb] = load_a_frag(xg, lane, kb);

    for (int jb = 0; jb < 8; ++jb) {
        const float lbv = lb[jb * 16 + (lane & 15)];
        v8f acc;
#pragma unroll
        for (int r = 0; r < 8; ++r) acc[r] = lbv;
#pragma unroll
        for (int kb = 0; kb < 4; ++kb)
            acc = wmma_f16(a0[kb], load_b_frag(wpack, 0, kb, jb, lane), acc);
        const int nn = jb * 16 + (lane & 15);
#pragma unroll
        for (int r = 0; r < 8; ++r) {
            const int m = r + ((lane < 16) ? 0 : 8);   // D layout: VGPR r, lane group
            xg[m * CCH + nn] = (_Float16)acc[r];
        }
    }

    // ---- stage 3: gpre = o0 @ gw^T + gb ; w1g = sigmoid(gpre) * fw1 ----
    v16h ag[4];
#pragma unroll
    for (int kb = 0; kb < 4; ++kb) ag[kb] = load_a_frag(xg, lane, kb);

    const float* fw1 = fw + CCH;                 // fw[1,0,:]
    for (int jb = 0; jb < 8; ++jb) {
        const float gbv = gb[jb * 16 + (lane & 15)];
        const float fwv = fw1[jb * 16 + (lane & 15)];
        v8f acc;
#pragma unroll
        for (int r = 0; r < 8; ++r) acc[r] = gbv;
#pragma unroll
        for (int kb = 0; kb < 4; ++kb)
            acc = wmma_f16(ag[kb], load_b_frag(wpack, 2, kb, jb, lane), acc);
#pragma unroll
        for (int r = 0; r < 8; ++r) {
            const float g = 1.f / (1.f + expf(-acc[r]));
            w1g[(jb * 8 + r) * 32 + lane] = g * fwv;
        }
    }

    // ---- stage 4: o1_l = (x_l*gate) @ lw1^T, fused with gating + fw1 dot ----
    float part[3][8];
#pragma unroll
    for (int l = 0; l < 3; ++l)
#pragma unroll
        for (int r = 0; r < 8; ++r) part[l][r] = 0.f;

    for (int jb = 0; jb < 8; ++jb) {
        v8f acc1, acc2, acc3;
#pragma unroll
        for (int r = 0; r < 8; ++r) { acc1[r] = 0.f; acc2[r] = 0.f; acc3[r] = 0.f; }
#pragma unroll
        for (int kb = 0; kb < 4; ++kb) {
            v16h b = load_b_frag(wpack, 1, kb, jb, lane);   // shared by 3 GEMMs
            acc1 = wmma_f16(load_a_frag(xg + 1 * (16 * CCH), lane, kb), b, acc1);
            acc2 = wmma_f16(load_a_frag(xg + 2 * (16 * CCH), lane, kb), b, acc2);
            acc3 = wmma_f16(load_a_frag(xg + 3 * (16 * CCH), lane, kb), b, acc3);
        }
#pragma unroll
        for (int r = 0; r < 8; ++r) {
            const float wg = w1g[(jb * 8 + r) * 32 + lane];
            part[0][r] += acc1[r] * wg;
            part[1][r] += acc2[r] * wg;
            part[2][r] += acc3[r] * wg;
        }
    }

    // ---- reduce over columns (16-lane groups) and store out[n, l] ----
#pragma unroll
    for (int l = 0; l < 3; ++l) {
#pragma unroll
        for (int r = 0; r < 8; ++r) {
            float v = part[l][r];
            v += __shfl_xor(v, 1, 32);
            v += __shfl_xor(v, 2, 32);
            v += __shfl_xor(v, 4, 32);
            v += __shfl_xor(v, 8, 32);
            const int m = r + ((lane < 16) ? 0 : 8);
            const int n = n0 + m;
            if ((lane & 15) == 0 && n < N) out[(size_t)n * 3 + l] = v;
        }
    }
}

// ---------------------------------------------------------------------------
extern "C" void kernel_launch(void* const* d_in, const int* in_sizes, int n_in,
                              void* d_out, int out_size, void* d_ws, size_t ws_size,
                              hipStream_t stream)
{
    const float* x     = (const float*)d_in[0];
    const float* t     = (const float*)d_in[1];
    const int*   bidx  = (const int*)  d_in[2];
    const float* tm_w1 = (const float*)d_in[3];
    const float* tm_b1 = (const float*)d_in[4];
    const float* tm_w2 = (const float*)d_in[5];
    const float* tm_b2 = (const float*)d_in[6];
    const float* lw    = (const float*)d_in[7];
    const float* lb    = (const float*)d_in[8];
    const float* gw    = (const float*)d_in[9];
    const float* gb    = (const float*)d_in[10];
    const float* fw    = (const float*)d_in[11];
    // d_in[12] (fb) is dead: output only uses l=1..3 rows (no fb term).
    float* out = (float*)d_out;

    const int N = in_sizes[0] / (9 * CCH);
    const int B = in_sizes[1];

    // workspace layout: [wpack: 3*128*128 f16 = 96KB][tb: B*128 f32][tg: B*128 f32]
    _Float16* wpack = (_Float16*)d_ws;
    float* tb = (float*)((char*)d_ws + (size_t)3 * CCH * CCH * sizeof(_Float16));
    float* tg = tb + (size_t)B * CCH;

    tmlp_kernel<<<B, 128, 0, stream>>>(t, tm_w1, tm_b1, tm_w2, tm_b2, tb, tg, B);
    pack_kernel<<<(3 * CCH * CCH + 255) / 256, 256, 0, stream>>>(lw, gw, wpack);

    const int ntiles  = (N + 15) / 16;
    const int nblocks = (ntiles + 3) / 4;
    const size_t shmem = 65536 + 32768;   // xg (4 waves x 16KB) + w1g (4 x 8KB)
    velocity_head_kernel<<<nblocks, 128, shmem, stream>>>(
        x, bidx, lb, gb, fw, wpack, tb, tg, out, N);
}
